// CGTEncoderLayer_18700287607049
// MI455X (gfx1250) — compile-verified, matrix operand order checked by reading, not run on hardware
//
#include <hip/hip_runtime.h>

#define BB 4
#define SS 1023
#define SP 1024
#define DM 1024
#define NH 16
#define DH 64
#define FFD 4096
#define MM (BB*SS)
#define NEGINF (-1e30f)

typedef __attribute__((ext_vector_type(16))) __bf16 v16bf;
typedef __attribute__((ext_vector_type(8)))  float  v8f;

// ---------- helpers ----------

static __device__ __forceinline__ unsigned short f2bf(float f) {
  unsigned int u = __float_as_uint(f);
  u += 0x7FFFu + ((u >> 16) & 1u);   // round-to-nearest-even
  return (unsigned short)(u >> 16);
}

// Build a 16x32 bf16 A/B-operand fragment for one lane from 32 consecutive
// bf16 values (CDNA5 16-bit A layout: VGPR j<4 -> k=2j+8*hi, j>=4 -> k=16+2(j-4)+8*hi,
// which is exactly dwords [4*hi..4*hi+3] and [8+4*hi..11+4*hi] of the row).
static __device__ __forceinline__ v16bf load_frag(const unsigned short* p32, int hi) {
  const uint4* p = (const uint4*)p32;
  union { uint4 u[2]; v16bf v; } c;
  c.u[0] = p[hi];
  c.u[1] = p[hi + 2];
  return c.v;
}

static __device__ __forceinline__ v8f wmma_bf16(v16bf a, v16bf b, v8f c) {
  return __builtin_amdgcn_wmma_f32_16x16x32_bf16(false, a, false, b, (short)0, c,
                                                 false, false);
}

static __device__ __forceinline__ float redmax16(float v) {
  v = fmaxf(v, __shfl_xor(v, 1, 32));
  v = fmaxf(v, __shfl_xor(v, 2, 32));
  v = fmaxf(v, __shfl_xor(v, 4, 32));
  v = fmaxf(v, __shfl_xor(v, 8, 32));
  return v;
}
static __device__ __forceinline__ float redsum16(float v) {
  v += __shfl_xor(v, 1, 32);
  v += __shfl_xor(v, 2, 32);
  v += __shfl_xor(v, 4, 32);
  v += __shfl_xor(v, 8, 32);
  return v;
}

// ---------- generic bf16-WMMA GEMM: C = act(A @ B + bias [+ Cin]) ----------
// A [M,K] fp32 row-major, B [K,N] fp32 row-major, C [M,N] fp32.
// act: 0 none, 1 relu, 2 sigmoid.  beta: add Cin before activation.
__global__ __launch_bounds__(256)
void gemm_kernel(const float* __restrict__ A, const float* __restrict__ Bm,
                 const float* __restrict__ bias, const float* __restrict__ Cin,
                 float* __restrict__ Cout, int Mdim, int Ndim, int Kdim,
                 int act, int beta) {
  __shared__ __align__(16) unsigned short As[2][128][40];  // rows of A (bf16)
  __shared__ __align__(16) unsigned short Bs[2][128][40];  // B transposed: [n][k]
  const int t = threadIdx.x;
  const int lane = t & 31, w = t >> 5;
  const int li = lane & 15, hi = lane >> 4;
  const int bm = blockIdx.y * 128, bn = blockIdx.x * 128;
  const int m0 = (w & 3) * 32, n0 = (w >> 2) * 64;

  v8f acc0[4], acc1[4];
#pragma unroll
  for (int u = 0; u < 4; ++u) { acc0[u] = (v8f)0.0f; acc1[u] = (v8f)0.0f; }

  const int nk = Kdim >> 5;

  auto load_tiles = [&](int kc, int buf) {
    const int kt = kc << 5;
#pragma unroll
    for (int i = 0; i < 4; ++i) {            // A: 128x32 fp32 -> bf16
      int f = t + (i << 8);
      int row = f >> 3, k4 = (f & 7) << 2;
      int gm = bm + row;
      float4 av = make_float4(0.f, 0.f, 0.f, 0.f);
      if (gm < Mdim) av = *(const float4*)(A + (size_t)gm * Kdim + kt + k4);
      union { unsigned short h[4]; uint2 u; } pk;
      pk.h[0] = f2bf(av.x); pk.h[1] = f2bf(av.y);
      pk.h[2] = f2bf(av.z); pk.h[3] = f2bf(av.w);
      *(uint2*)&As[buf][row][k4] = pk.u;
    }
#pragma unroll
    for (int i = 0; i < 4; ++i) {            // B: 32x128 fp32 -> bf16, transposed
      int f = t + (i << 8);
      int kr = f >> 5, n4 = (f & 31) << 2;
      const float* bp = Bm + (size_t)(kt + kr) * Ndim;
#pragma unroll
      for (int j = 0; j < 4; ++j) {
        int gn = bn + n4 + j;
        float v = (gn < Ndim) ? bp[gn] : 0.f;
        Bs[buf][n4 + j][kr] = f2bf(v);
      }
    }
  };

  load_tiles(0, 0);
  __syncthreads();
  for (int kc = 0; kc < nk; ++kc) {
    const int cur = kc & 1;
    if (kc + 1 < nk) load_tiles(kc + 1, cur ^ 1);
    if (kc + 2 < nk) {                       // light L2 prefetch of A
      int row = t >> 3, k4 = (t & 7) << 2;
      int gm = bm + row;
      if (gm < Mdim) __builtin_prefetch(A + (size_t)gm * Kdim + ((kc + 2) << 5) + k4, 0, 0);
    }
    v16bf a0 = load_frag(&As[cur][m0 + li][0], hi);
    v16bf a1 = load_frag(&As[cur][m0 + 16 + li][0], hi);
#pragma unroll
    for (int u = 0; u < 4; ++u) {
      v16bf bu = load_frag(&Bs[cur][n0 + (u << 4) + li][0], hi);
      acc0[u] = wmma_bf16(a0, bu, acc0[u]);
      acc1[u] = wmma_bf16(a1, bu, acc1[u]);
    }
    __syncthreads();
  }

  // epilogue: C layout -> VGPR r holds rows r (lanes 0-15) / r+8 (lanes 16-31)
#pragma unroll
  for (int i = 0; i < 2; ++i) {
#pragma unroll
    for (int u = 0; u < 4; ++u) {
      v8f a = i ? acc1[u] : acc0[u];
#pragma unroll
      for (int r = 0; r < 8; ++r) {
        int gm = bm + m0 + (i << 4) + r + (hi << 3);
        int gn = bn + n0 + (u << 4) + li;
        if (gm < Mdim && gn < Ndim) {
          float v = a[r];
          if (bias) v += bias[gn];
          if (beta) v += Cin[(size_t)gm * Ndim + gn];
          if (act == 1) v = fmaxf(v, 0.f);
          else if (act == 2) v = 1.f / (1.f + __expf(-v));
          Cout[(size_t)gm * Ndim + gn] = v;
        }
      }
    }
  }
}

// ---------- conversion kernels ----------
// fp32 [B,S,DM] -> bf16 [B,SP,DM], rows >= S zero-padded
__global__ __launch_bounds__(256)
void cvt_rows_bf16(const float* __restrict__ src, unsigned short* __restrict__ dst) {
  size_t idx = (size_t)blockIdx.x * 256 + threadIdx.x;
  int col = (int)(idx & (DM - 1));
  int pr = (int)(idx >> 10);
  int b = pr >> 10, s = pr & (SP - 1);
  float v = (s < SS) ? src[((size_t)(b * SS + s)) * DM + col] : 0.f;
  dst[idx] = f2bf(v);
}

// V fp32 [B,S,H*64] -> bf16 [B,H,64,SP] (d-major, key-contiguous), padded keys zero
__global__ __launch_bounds__(256)
void cvt_vT_bf16(const float* __restrict__ src, unsigned short* __restrict__ dst) {
  size_t idx = (size_t)blockIdx.x * 256 + threadIdx.x;
  int sp = (int)(idx & (SP - 1));
  int d  = (int)((idx >> 10) & 63);
  int h  = (int)((idx >> 16) & (NH - 1));
  int b  = (int)(idx >> 20);
  float v = (sp < SS) ? src[((size_t)(b * SS + sp)) * DM + h * DH + d] : 0.f;
  dst[idx] = f2bf(v);
}

// ---------- flash attention with WMMA (1 wave / 16 query rows / head) ----------
__global__ __launch_bounds__(32)
void attn_kernel(const unsigned short* __restrict__ qb,
                 const unsigned short* __restrict__ kb,
                 const unsigned short* __restrict__ vt,
                 const float* __restrict__ rel_emb,
                 const float* __restrict__ ebias,
                 const int* __restrict__ smask,
                 float* __restrict__ ctx) {
  const int bh = blockIdx.x, b = bh / NH, h = bh % NH;
  const int q0 = blockIdx.y * 16;
  const int lane = threadIdx.x, li = lane & 15, hi = lane >> 4;
  __shared__ __align__(16) unsigned short Ps[16][32];

  const unsigned short* qrow = qb + ((size_t)b * SP + q0 + li) * DM + h * DH;
  const v16bf aq0 = load_frag(qrow, hi);        // d 0..31
  const v16bf aq1 = load_frag(qrow + 32, hi);   // d 32..63

  v8f o[4];
  float rmax[8], rsum[8];
#pragma unroll
  for (int u = 0; u < 4; ++u) o[u] = (v8f)0.0f;
#pragma unroll
  for (int r = 0; r < 8; ++r) { rmax[r] = NEGINF; rsum[r] = 0.f; }

  for (int kt = 0; kt < SP; kt += 32) {
    // ---- scores: Q(16x64) @ K^T(64x32) ----
    v8f st[2];
#pragma unroll
    for (int tc = 0; tc < 2; ++tc) {
      const unsigned short* krow = kb + ((size_t)b * SP + kt + tc * 16 + li) * DM + h * DH;
      v8f c = (v8f)0.0f;
      c = wmma_bf16(aq0, load_frag(krow, hi), c);
      c = wmma_bf16(aq1, load_frag(krow + 32, hi), c);
      st[tc] = c;
    }
    // ---- bias + mask + online softmax ----
#pragma unroll
    for (int r = 0; r < 8; ++r) {
      const int qi = q0 + r + (hi << 3);
      float sv[2];
#pragma unroll
      for (int tc = 0; tc < 2; ++tc) {
        const int ki = kt + tc * 16 + li;
        float s = st[tc][r] * 0.125f;                 // 1/sqrt(64)
        if (qi < SS && ki < SS) {
          s += rel_emb[(qi - ki + (SP - 1)) * NH + h];
          if (ki == qi + 2 && (qi & 1) == 0)
            s += ebias[((size_t)(b * SS + qi + 1)) * NH + h];
          if (smask[b * SS + ki] == 0) s = NEGINF;
        } else s = NEGINF;
        sv[tc] = s;
      }
      float mx = redmax16(fmaxf(sv[0], sv[1]));
      float nm = fmaxf(rmax[r], mx);
      float corr = __expf(rmax[r] - nm);
      rmax[r] = nm;
      float p0 = __expf(sv[0] - nm), p1 = __expf(sv[1] - nm);
      rsum[r] = rsum[r] * corr + redsum16(p0 + p1);
#pragma unroll
      for (int u = 0; u < 4; ++u) o[u][r] *= corr;
      Ps[r + (hi << 3)][li]      = f2bf(p0);
      Ps[r + (hi << 3)][16 + li] = f2bf(p1);
    }
    __syncthreads();
    // ---- O += P(16x32) @ V(32x64), V pre-transposed so cols are contiguous ----
    v16bf ap = load_frag(&Ps[li][0], hi);
#pragma unroll
    for (int u = 0; u < 4; ++u) {
      const unsigned short* vrow =
          vt + (((size_t)(b * NH + h)) * DH + u * 16 + li) * SP + kt;
      o[u] = wmma_bf16(ap, load_frag(vrow, hi), o[u]);
    }
    __syncthreads();
  }

#pragma unroll
  for (int u = 0; u < 4; ++u)
#pragma unroll
    for (int r = 0; r < 8; ++r) {
      int qi = q0 + r + (hi << 3);
      if (qi < SS)
        ctx[((size_t)(b * SS + qi)) * DM + h * DH + u * 16 + li] = o[u][r] / rsum[r];
    }
}

// ---------- dual causal probe: rowwise max of strict-lower sims ----------
__global__ __launch_bounds__(256)
void probe_kernel(const unsigned short* __restrict__ ppb,
                  const unsigned short* __restrict__ phb,
                  const unsigned short* __restrict__ kcb,
                  float* __restrict__ spre, float* __restrict__ shaz) {
  const int b = blockIdx.x >> 6, qt = blockIdx.x & 63;
  const int q0 = qt * 16;
  const int t = threadIdx.x, lane = t & 31, w = t >> 5;
  const int li = lane & 15, hi = lane >> 4;
  __shared__ float redp[8][16], redh[8][16];

  const unsigned short* prow = ppb + ((size_t)b * SP + q0 + li) * DM;
  const unsigned short* hrow = phb + ((size_t)b * SP + q0 + li) * DM;

  float mp[8], mh[8];
#pragma unroll
  for (int r = 0; r < 8; ++r) { mp[r] = NEGINF; mh[r] = NEGINF; }

  for (int kti = w; kti <= qt; kti += 8) {
    const int kbase = kti * 16;
    const unsigned short* krow = kcb + ((size_t)b * SP + kbase + li) * DM;
    v8f cp = (v8f)0.0f, ch = (v8f)0.0f;
    for (int c = 0; c < 32; ++c) {            // K = 1024 in 32-chunks
      v16bf bk = load_frag(krow + c * 32, hi);
      cp = wmma_bf16(load_frag(prow + c * 32, hi), bk, cp);
      ch = wmma_bf16(load_frag(hrow + c * 32, hi), bk, ch);
    }
#pragma unroll
    for (int r = 0; r < 8; ++r) {
      const int qi = q0 + r + (hi << 3);
      const int ki = kbase + li;
      float sp = cp[r] * 0.03125f;            // 1/sqrt(1024)
      float sh = ch[r] * 0.03125f;
      const bool ok = (ki < qi) && (qi < SS);
      if (!ok) { sp = NEGINF; sh = NEGINF; }
      mp[r] = fmaxf(mp[r], redmax16(sp));
      mh[r] = fmaxf(mh[r], redmax16(sh));
    }
  }
  if (li == 0) {
#pragma unroll
    for (int r = 0; r < 8; ++r) {
      redp[w][r + (hi << 3)] = mp[r];
      redh[w][r + (hi << 3)] = mh[r];
    }
  }
  __syncthreads();
  if (t < 16) {
    float a = NEGINF, c = NEGINF;
#pragma unroll
    for (int ww = 0; ww < 8; ++ww) { a = fmaxf(a, redp[ww][t]); c = fmaxf(c, redh[ww][t]); }
    const int qi = q0 + t;
    if (qi < SS) {
      spre[b * SS + qi] = (a < -9e29f) ? 0.f : a;
      shaz[b * SS + qi] = (c < -9e29f) ? 0.f : c;
    }
  }
}

// ---------- SE MLP of the causal gate unit ----------
__global__ __launch_bounds__(128)
void se_kernel(const float* __restrict__ gates, const float* __restrict__ spre,
               const float* __restrict__ shaz,
               const float* __restrict__ se1_w, const float* __restrict__ se1_b,
               const float* __restrict__ se2_w, const float* __restrict__ se2_b,
               float* __restrict__ h_logic) {
  const int tok = blockIdx.x;
  __shared__ float hid[512];
  const float r0 = gates[tok * 2 + 0], r2 = gates[tok * 2 + 1];
  const float r1 = spre[tok], r3 = shaz[tok];
  const int t = threadIdx.x;
  for (int j = t; j < 512; j += 128) {
    float v = se1_b[j] + r0 * se1_w[j] + r1 * se1_w[512 + j] +
              r2 * se1_w[1024 + j] + r3 * se1_w[1536 + j];
    hid[j] = fmaxf(v, 0.f);
  }
  __syncthreads();
  for (int j = t; j < DM; j += 128) {
    float acc = se2_b[j];
    for (int k = 0; k < 512; ++k) acc += hid[k] * se2_w[(size_t)k * DM + j];
    h_logic[(size_t)tok * DM + j] = acc;
  }
}

// ---------- fused residual + (optional gate mix) + LayerNorm ----------
__global__ __launch_bounds__(256)
void residual_ln_kernel(const float* __restrict__ xin, const float* __restrict__ a1,
                        const float* __restrict__ a2, const float* __restrict__ g,
                        const float* __restrict__ gam, const float* __restrict__ bet,
                        float* __restrict__ outp) {
  const int tok = blockIdx.x, t = threadIdx.x;
  const size_t base = (size_t)tok * DM;
  __shared__ float sm[256];
  float vals[4];
  float s = 0.f;
#pragma unroll
  for (int i = 0; i < 4; ++i) {
    int j = t + (i << 8);
    float tv;
    if (g) {
      float gv = g[base + j];
      tv = xin[base + j] + gv * a1[base + j] + (1.f - gv) * a2[base + j];
    } else {
      tv = xin[base + j] + a1[base + j];
    }
    vals[i] = tv; s += tv;
  }
  sm[t] = s; __syncthreads();
  for (int off = 128; off > 0; off >>= 1) {
    if (t < off) sm[t] += sm[t + off];
    __syncthreads();
  }
  const float mu = sm[0] * (1.f / DM);
  __syncthreads();
  float s2 = 0.f;
#pragma unroll
  for (int i = 0; i < 4; ++i) { float d = vals[i] - mu; s2 += d * d; }
  sm[t] = s2; __syncthreads();
  for (int off = 128; off > 0; off >>= 1) {
    if (t < off) sm[t] += sm[t + off];
    __syncthreads();
  }
  const float inv = rsqrtf(sm[0] * (1.f / DM) + 1e-5f);
#pragma unroll
  for (int i = 0; i < 4; ++i) {
    int j = t + (i << 8);
    outp[base + j] = (vals[i] - mu) * inv * gam[j] + bet[j];
  }
}

// ---------- orchestration ----------
static inline void launch_gemm(const float* A, const float* Bm, const float* bias,
                               const float* Cin, float* Cout, int M_, int N_, int K_,
                               int act, int beta, hipStream_t s) {
  dim3 g((N_ + 127) / 128, (M_ + 127) / 128);
  gemm_kernel<<<g, 256, 0, s>>>(A, Bm, bias, Cin, Cout, M_, N_, K_, act, beta);
}

extern "C" void kernel_launch(void* const* d_in, const int* in_sizes, int n_in,
                              void* d_out, int out_size, void* d_ws, size_t ws_size,
                              hipStream_t stream) {
  (void)in_sizes; (void)n_in; (void)out_size; (void)ws_size;
  const float* x     = (const float*)d_in[0];
  const float* temb  = (const float*)d_in[2];
  const int*   smask = (const int*)d_in[3];
  const float *sq_w = (const float*)d_in[4],  *sq_b = (const float*)d_in[5];
  const float *sk_w = (const float*)d_in[6],  *sk_b = (const float*)d_in[7];
  const float *sv_w = (const float*)d_in[8],  *sv_b = (const float*)d_in[9];
  const float *so_w = (const float*)d_in[10], *so_b = (const float*)d_in[11];
  const float *rel  = (const float*)d_in[12];
  const float *edge_w = (const float*)d_in[13];
  const float *q_w  = (const float*)d_in[14], *q_b  = (const float*)d_in[15];
  const float *k_w  = (const float*)d_in[16], *k_b  = (const float*)d_in[17];
  const float *pre_w = (const float*)d_in[18], *pre_b = (const float*)d_in[19];
  const float *haz_w = (const float*)d_in[20], *haz_b = (const float*)d_in[21];
  const float *gate_w = (const float*)d_in[22], *gate_b = (const float*)d_in[23];
  const float *se1_w = (const float*)d_in[24], *se1_b = (const float*)d_in[25];
  const float *se2_w = (const float*)d_in[26], *se2_b = (const float*)d_in[27];
  const float *fus_w = (const float*)d_in[28], *fus_b = (const float*)d_in[29];
  const float *ff1_w = (const float*)d_in[30], *ff1_b = (const float*)d_in[31];
  const float *ff2_w = (const float*)d_in[32], *ff2_b = (const float*)d_in[33];
  const float *ln1_g = (const float*)d_in[34], *ln1_b = (const float*)d_in[35];
  const float *ln2_g = (const float*)d_in[36], *ln2_b = (const float*)d_in[37];

  char* wp = (char*)d_ws; size_t off = 0;
  auto take = [&](size_t bytes) -> void* {
    void* p = wp + off; off += (bytes + 255) & ~(size_t)255; return p;
  };
  const size_t MDf = (size_t)MM * DM * sizeof(float);
  float* fA  = (float*)take(MDf);                              // q / ctx / kc / p_* / h_logic
  float* fB  = (float*)take(MDf);                              // k / out_ctx / ff2out
  float* fC  = (float*)take(MDf);                              // v / qc / g_fus
  float* fD  = (float*)take(MDf);                              // x1
  float* fFF = (float*)take((size_t)MM * FFD * sizeof(float)); // FFN hidden
  float* ebias = (float*)take((size_t)MM * NH * sizeof(float));
  float* gates = (float*)take((size_t)MM * 2 * sizeof(float));
  float* spre  = (float*)take((size_t)MM * sizeof(float));
  float* shaz  = (float*)take((size_t)MM * sizeof(float));
  const size_t BFb = (size_t)BB * SP * DM * sizeof(unsigned short);
  unsigned short* qb  = (unsigned short*)take(BFb);
  unsigned short* kb  = (unsigned short*)take(BFb);
  unsigned short* vt  = (unsigned short*)take(BFb);
  unsigned short* kcb = (unsigned short*)take(BFb);
  unsigned short* ppb = (unsigned short*)take(BFb);
  unsigned short* phb = (unsigned short*)take(BFb);

  const int cvtBlocks = (BB * SP * DM) / 256;

  // structural attention: projections -> bf16 -> flash attention -> out proj
  launch_gemm(x, sq_w, sq_b, nullptr, fA, MM, DM, DM, 0, 0, stream);
  launch_gemm(x, sk_w, sk_b, nullptr, fB, MM, DM, DM, 0, 0, stream);
  launch_gemm(x, sv_w, sv_b, nullptr, fC, MM, DM, DM, 0, 0, stream);
  cvt_rows_bf16<<<cvtBlocks, 256, 0, stream>>>(fA, qb);
  cvt_rows_bf16<<<cvtBlocks, 256, 0, stream>>>(fB, kb);
  cvt_vT_bf16<<<cvtBlocks, 256, 0, stream>>>(fC, vt);
  launch_gemm(temb, edge_w, nullptr, nullptr, ebias, MM, NH, DM, 0, 0, stream);
  attn_kernel<<<dim3(BB * NH, SP / 16), 32, 0, stream>>>(qb, kb, vt, rel, ebias,
                                                         smask, fA);          // ctx
  launch_gemm(fA, so_w, so_b, nullptr, fB, MM, DM, DM, 0, 0, stream);         // out_ctx

  // causal gate unit
  launch_gemm(x, q_w, q_b, nullptr, fC, MM, DM, DM, 0, 0, stream);            // qc
  launch_gemm(x, k_w, k_b, nullptr, fA, MM, DM, DM, 0, 0, stream);            // kc
  cvt_rows_bf16<<<cvtBlocks, 256, 0, stream>>>(fA, kcb);
  launch_gemm(fC, pre_w, pre_b, nullptr, fA, MM, DM, DM, 0, 0, stream);       // p_pre
  cvt_rows_bf16<<<cvtBlocks, 256, 0, stream>>>(fA, ppb);
  launch_gemm(fC, haz_w, haz_b, nullptr, fA, MM, DM, DM, 0, 0, stream);       // p_haz
  cvt_rows_bf16<<<cvtBlocks, 256, 0, stream>>>(fA, phb);
  launch_gemm(fC, gate_w, gate_b, nullptr, gates, MM, 2, DM, 2, 0, stream);   // gates
  probe_kernel<<<BB * 64, 256, 0, stream>>>(ppb, phb, kcb, spre, shaz);
  se_kernel<<<MM, 128, 0, stream>>>(gates, spre, shaz, se1_w, se1_b, se2_w,
                                    se2_b, fA);                               // h_logic

  // fusion gate: sigmoid(out_ctx @ Wtop + h_logic @ Wbot + b)
  launch_gemm(fB, fus_w, fus_b, nullptr, fC, MM, DM, DM, 0, 0, stream);
  launch_gemm(fA, fus_w + (size_t)DM * DM, nullptr, fC, fC, MM, DM, DM, 2, 1, stream);

  // x1 = LN(x + g*out_ctx + (1-g)*h_logic)
  residual_ln_kernel<<<MM, 256, 0, stream>>>(x, fB, fA, fC, ln1_g, ln1_b, fD);

  // FFN + final LN
  launch_gemm(fD, ff1_w, ff1_b, nullptr, fFF, MM, FFD, DM, 1, 0, stream);
  launch_gemm(fFF, ff2_w, ff2_b, nullptr, fB, MM, DM, FFD, 0, 0, stream);
  residual_ln_kernel<<<MM, 256, 0, stream>>>(fD, fB, nullptr, nullptr, ln2_g,
                                             ln2_b, (float*)d_out);
}
